// A2CNet_18837726560735
// MI455X (gfx1250) — matrix operand order repaired and checked by reading, not compiled, not run on hardware
//
#include <hip/hip_runtime.h>
#include <hip/hip_bf16.h>

typedef __attribute__((ext_vector_type(16))) _Float16 v16h;
typedef __attribute__((ext_vector_type(8)))  _Float16 v8h;
typedef __attribute__((ext_vector_type(8)))  float    v8f;

#define TSTEPS 128

__device__ __forceinline__ float hsig(float v) {
    float r = 0.2f * v + 0.5f;
    return fminf(fmaxf(r, 0.0f), 1.0f);
}

__device__ __forceinline__ v8h zero8h() {
    v8h z;
    #pragma unroll
    for (int i = 0; i < 8; ++i) z[i] = (_Float16)0.0f;
    return z;
}

// ---------------------------------------------------------------------------
// gfx1250 async global->LDS copy (CDNA5 ISA, VGLOBAL op 98).
// vdst = VGPR holding the 32-bit LDS byte offset, vaddr = 64-bit global addr.
// Tracked by ASYNCcnt; must s_wait_asynccnt before consuming the LDS data.
// Generic pointers to __shared__ carry the LDS offset in their low 32 bits.
// ---------------------------------------------------------------------------
__device__ __forceinline__ void async_copy_b128(void* lds_ptr, const void* gptr) {
    unsigned lds_off = (unsigned)(size_t)lds_ptr;
    asm volatile("global_load_async_to_lds_b128 %0, %1, off"
                 :: "v"(lds_off), "v"(gptr) : "memory");
}
__device__ __forceinline__ void wait_async() {
    asm volatile("s_wait_asynccnt 0x0" ::: "memory");
}

// ---------------------------------------------------------------------------
// Cooperative halo staging, CIN=32: rows y-1..y+1, cols x0-1..x0+16 (18 px),
// 32 fp16 channels -> halo[3][20][32] (stride-padded).  216 b128 chunks.
// ---------------------------------------------------------------------------
__device__ __forceinline__ void stage_halo32(_Float16* halo /* [3][20][32] */,
                                             const _Float16* __restrict__ img,
                                             int y, int x0, int tid) {
    if (tid < 216) {
        const int r   = tid / 72;
        const int rem = tid - r * 72;
        const int e   = rem >> 2;          // 0..17
        const int cg  = rem & 3;           // 8-channel chunk
        const int iy  = y + r - 1;
        const int ix  = x0 + e - 1;
        _Float16* dst = halo + ((r * 20) + e) * 32 + cg * 8;
        if ((unsigned)iy < 64u && (unsigned)ix < 64u) {
            async_copy_b128(dst, img + (size_t)((iy << 6) + ix) * 32 + cg * 8);
        } else {
            *(v8h*)dst = zero8h();
        }
    }
    wait_async();
    __syncthreads();
}

// ---------------------------------------------------------------------------
// 16x16 WMMA conv tile, CIN=32 (K=288 = 9 taps x 32ch).  A from LDS halo
// (unconditional ds_load_b128), B from packed global weights [Cout][288].
// A-fragment (ISA 7.12.2, 16-bit A 16x32): lane l (hi=l>>4), elements
// [g*8,g*8+8) = K kt*32 + g*16 + hi*8 + [0..8)  ->  channel c = g*16+hi*8.
// ---------------------------------------------------------------------------
__device__ __forceinline__ v8f conv_tile32(const _Float16* halo /* [3][20][32] */,
                                           const _Float16* __restrict__ wcol,
                                           int l16, int hi, v8f acc) {
    #pragma unroll
    for (int kt = 0; kt < 9; ++kt) {
        const int ky = kt / 3, kx = kt % 3;
        union { v16h v; v8h h[2]; } A, B;
        const _Float16* hp = halo + ((ky * 20) + (l16 + kx)) * 32;
        A.h[0] = *(const v8h*)(hp + hi * 8);
        A.h[1] = *(const v8h*)(hp + 16 + hi * 8);
        if (kt + 1 < 9)
            __builtin_prefetch(wcol + (kt + 1) * 32, 0, 1);   // global_prefetch_b8
        B.h[0] = *(const v8h*)(wcol + kt * 32 + hi * 8);
        B.h[1] = *(const v8h*)(wcol + kt * 32 + 16 + hi * 8);
        acc = __builtin_amdgcn_wmma_f32_16x16x32_f16(
                  false, A.v, false, B.v, (short)0, acc, false, false);
    }
    return acc;
}

// ---------------------------------------------------------------------------
// CIN=16 variant (K=144 padded to 160).  halo[3][68][16] covers a whole row
// (64 px + borders).  tap = kt*2+g (compile-time); tap 9 is the zero pad.
// ---------------------------------------------------------------------------
__device__ __forceinline__ v8f conv_tile16(const _Float16* halo /* [3][68][16] */,
                                           const _Float16* __restrict__ wcol,
                                           int xloc, int hi, v8f acc) {
    #pragma unroll
    for (int kt = 0; kt < 5; ++kt) {
        union { v16h v; v8h h[2]; } A, B;
        #pragma unroll
        for (int g = 0; g < 2; ++g) {
            const int tap = kt * 2 + g;
            if (tap < 9) {
                const int ky = tap / 3, kx = tap % 3;
                A.h[g] = *(const v8h*)(halo + ((ky * 68) + (xloc + kx)) * 16 + hi * 8);
            } else {
                A.h[g] = zero8h();
            }
        }
        if (kt + 1 < 5)
            __builtin_prefetch(wcol + (kt + 1) * 32, 0, 1);
        B.h[0] = *(const v8h*)(wcol + kt * 32 + hi * 8);
        B.h[1] = *(const v8h*)(wcol + kt * 32 + 16 + hi * 8);
        acc = __builtin_amdgcn_wmma_f32_16x16x32_f16(
                  false, A.v, false, B.v, (short)0, acc, false, false);
    }
    return acc;
}

// ---------------------------------------------------------------------------
// Pack weights HWIO -> [Cout][Kpad] fp16 (zero pad), zero c0 and h0.
// ---------------------------------------------------------------------------
__global__ void init_pack(const float* __restrict__ w2, const float* __restrict__ wx,
                          const float* __restrict__ wh,
                          _Float16* __restrict__ w2p, _Float16* __restrict__ wxp,
                          _Float16* __restrict__ whp,
                          float* __restrict__ c0, _Float16* __restrict__ hh0) {
    const int stride = gridDim.x * blockDim.x;
    for (int i = blockIdx.x * blockDim.x + threadIdx.x; i < 131072; i += stride) {
        if (i < 32 * 160) {
            int cout = i / 160, k = i - cout * 160;         // k = tap*16 + c
            w2p[i] = (_Float16)((k < 144) ? w2[k * 32 + cout] : 0.0f);
        }
        if (i < 128 * 288) {
            int cout = i / 288, k = i - cout * 288;         // k = tap*32 + c
            wxp[i] = (_Float16)wx[k * 128 + cout];
            whp[i] = (_Float16)wh[k * 128 + cout];
        }
        c0[i]  = 0.0f;
        hh0[i] = (_Float16)0.0f;
    }
}

// ---------------------------------------------------------------------------
// Fused transpose + one-hot + 1x1 embed + 5x5 conv (1->16) + ReLU (VALU-only,
// ~1% of FLOPs).  x1 fp16 [T][64][64][16].
// ---------------------------------------------------------------------------
__global__ void embed_conv1(const int* __restrict__ states,
                            const float* __restrict__ wemb, const float* __restrict__ bemb,
                            const float* __restrict__ w1,   const float* __restrict__ b1,
                            _Float16* __restrict__ x1) {
    const int gid = blockIdx.x * blockDim.x + threadIdx.x;
    if (gid >= TSTEPS * 4096) return;
    const int t = gid >> 12, p = gid & 4095;
    const int y = p >> 6,    x = p & 63;
    const int* st = states + (size_t)t * 4096;
    float acc[16];
    #pragma unroll
    for (int c = 0; c < 16; ++c) acc[c] = b1[c];
    for (int ky = 0; ky < 5; ++ky) {
        const int iy = y + ky - 2;
        if ((unsigned)iy >= 64u) continue;
        for (int kx = 0; kx < 5; ++kx) {
            const int ix = x + kx - 2;
            if ((unsigned)ix >= 64u) continue;
            const int s = st[ix * 64 + iy];               // transpose(0,2,1)
            const float e = wemb[s] + bemb[0];            // one-hot @ 1x1 conv
            const float* wr = w1 + (ky * 5 + kx) * 16;
            #pragma unroll
            for (int c = 0; c < 16; ++c) acc[c] += e * wr[c];
        }
    }
    _Float16* out = x1 + (size_t)gid * 16;
    #pragma unroll
    for (int c = 0; c < 16; ++c) out[c] = (_Float16)fmaxf(acc[c], 0.0f);
}

// ---------------------------------------------------------------------------
// 3x3 conv 16->32 + bias + ReLU via WMMA.  One block = one image row
// (4 M-tiles x 2 N-tiles = 8 waves); halo for the whole row staged async.
// ---------------------------------------------------------------------------
__global__ void __launch_bounds__(256)
conv2_wmma(const _Float16* __restrict__ x1, const _Float16* __restrict__ w2p,
           const float* __restrict__ b2, _Float16* __restrict__ x2) {
    __shared__ _Float16 halo[3 * 68 * 16];
    const int t = blockIdx.x >> 6;
    const int y = blockIdx.x & 63;
    const int tid = threadIdx.x;
    const _Float16* img = x1 + (size_t)t * 4096 * 16;

    // stage rows y-1..y+1, cols -1..64 (66 px), 16 ch: 396 b128 chunks
    for (int idx = tid; idx < 396; idx += 256) {
        const int r   = idx / 132;
        const int rem = idx - r * 132;
        const int e   = rem >> 1;           // 0..65
        const int cg  = rem & 1;
        const int iy  = y + r - 1;
        const int ix  = e - 1;
        _Float16* dst = halo + ((r * 68) + e) * 16 + cg * 8;
        if ((unsigned)iy < 64u && (unsigned)ix < 64u)
            async_copy_b128(dst, img + (size_t)((iy << 6) + ix) * 16 + cg * 8);
        else
            *(v8h*)dst = zero8h();
    }
    wait_async();
    __syncthreads();

    const int w     = tid >> 5;             // wave 0..7
    const int ntile = w & 1;
    const int msub  = w >> 1;               // 0..3: strip within the row
    const int lid   = tid & 31, l16 = lid & 15, hi = lid >> 4;
    const int xloc  = msub * 16 + l16;      // halo column (+kx later)

    v8f acc = {};
    acc = conv_tile16(halo, w2p + (size_t)(ntile * 16 + l16) * 160, xloc, hi, acc);

    const int n = ntile * 16 + l16;
    const float bias = b2[n];
    _Float16* out = x2 + (size_t)t * 4096 * 32;
    #pragma unroll
    for (int r = 0; r < 8; ++r) {
        const int p = (y * 64) + msub * 16 + hi * 8 + r;  // D layout: M = r+hi*8
        out[(size_t)p * 32 + n] = (_Float16)fmaxf(acc[r] + bias, 0.0f);
    }
}

// ---------------------------------------------------------------------------
// 3x3 conv 32->128 (+b_lstm) via WMMA, parallel over T.  One block = one
// 16-pixel M-tile x 8 N-tile waves.  xz f32 [T][64][64][128].
// ---------------------------------------------------------------------------
__global__ void __launch_bounds__(256)
convx_wmma(const _Float16* __restrict__ x2, const _Float16* __restrict__ wxp,
           const float* __restrict__ blstm, float* __restrict__ xz) {
    __shared__ _Float16 halo[3 * 20 * 32];
    const int t     = blockIdx.x >> 8;
    const int mtile = blockIdx.x & 255;
    const int y     = mtile >> 2;
    const int x0    = (mtile & 3) << 4;
    const _Float16* img = x2 + (size_t)t * 4096 * 32;

    stage_halo32(halo, img, y, x0, threadIdx.x);

    const int ntile = threadIdx.x >> 5;
    const int lid = threadIdx.x & 31, l16 = lid & 15, hi = lid >> 4;
    v8f acc = {};
    acc = conv_tile32(halo, wxp + (size_t)(ntile * 16 + l16) * 288, l16, hi, acc);

    const int n = ntile * 16 + l16;
    const float bias = blstm[n];
    float* out = xz + (size_t)t * 4096 * 128;
    #pragma unroll
    for (int r = 0; r < 8; ++r) {
        const int p = mtile * 16 + hi * 8 + r;
        out[(size_t)p * 128 + n] = acc[r] + bias;
    }
}

// ---------------------------------------------------------------------------
// One fused ConvLSTM step: async-stage h halo -> WMMA conv3x3(h, Wh) ->
// zt exchange via LDS -> gates -> c/h update.  h double-buffered.
// ---------------------------------------------------------------------------
__global__ void __launch_bounds__(256)
lstm_step(const _Float16* __restrict__ hh_in, const _Float16* __restrict__ whp,
          const float* __restrict__ xz_t, float* __restrict__ c,
          float* __restrict__ hf, _Float16* __restrict__ hh_out) {
    __shared__ _Float16 halo[3 * 20 * 32];
    __shared__ float zt[16 * 128];
    const int mtile = blockIdx.x;              // 0..255
    const int y  = mtile >> 2;
    const int x0 = (mtile & 3) << 4;

    stage_halo32(halo, hh_in, y, x0, threadIdx.x);

    const int ntile = threadIdx.x >> 5;        // wave id = N-tile
    const int lid = threadIdx.x & 31, l16 = lid & 15, hi = lid >> 4;
    v8f acc = {};
    acc = conv_tile32(halo, whp + (size_t)(ntile * 16 + l16) * 288, l16, hi, acc);

    const int n = ntile * 16 + l16;
    #pragma unroll
    for (int r = 0; r < 8; ++r) {
        const int m = hi * 8 + r;
        const int p = mtile * 16 + m;
        zt[m * 128 + n] = acc[r] + xz_t[(size_t)p * 128 + n];
    }
    __syncthreads();

    // 16 pixels x 32 channels, keras gate order i,f,g,o along channels.
    for (int e = threadIdx.x; e < 512; e += 256) {
        const int m = e >> 5, ch = e & 31;
        const float iv = hsig(zt[m * 128 + ch]);
        const float fv = hsig(zt[m * 128 + ch + 32]);
        const float gv = tanhf(zt[m * 128 + ch + 64]);
        const float ov = hsig(zt[m * 128 + ch + 96]);
        const size_t idx = (size_t)(mtile * 16 + m) * 32 + ch;
        const float cn = fv * c[idx] + iv * gv;
        c[idx] = cn;
        const float hn = ov * tanhf(cn);
        hf[idx] = hn;
        hh_out[idx] = (_Float16)hn;
    }
}

// ---------------------------------------------------------------------------
// Heads.
// ---------------------------------------------------------------------------
__global__ void spatial_head(const float* __restrict__ hf, const float* __restrict__ ws,
                             const float* __restrict__ bs, float* __restrict__ sf) {
    const int p = blockIdx.x * blockDim.x + threadIdx.x;
    if (p >= 4096) return;
    float a = bs[0];
    const float* hp = hf + (size_t)p * 32;
    #pragma unroll
    for (int cch = 0; cch < 32; ++cch) a += hp[cch] * ws[cch];
    sf[p] = a;
}

__global__ void __launch_bounds__(256)
policy_gemv(const float* __restrict__ sf,
            const float* __restrict__ wpx, const float* __restrict__ bpx,
            const float* __restrict__ wpy, const float* __restrict__ bpy,
            float* __restrict__ logits) {
    __shared__ float red[256];
    const int head = blockIdx.x >> 6;
    const int j    = blockIdx.x & 63;
    const float* W = head ? wpy : wpx;
    const float* B = head ? bpy : bpx;
    float a = 0.0f;
    for (int p = threadIdx.x; p < 4096; p += 256) a += sf[p] * W[(size_t)p * 64 + j];
    red[threadIdx.x] = a; __syncthreads();
    for (int s = 128; s > 0; s >>= 1) {
        if (threadIdx.x < s) red[threadIdx.x] += red[threadIdx.x + s];
        __syncthreads();
    }
    if (threadIdx.x == 0) logits[blockIdx.x] = red[0] + B[j];
}

__global__ void softmax64(const float* __restrict__ logits, float* __restrict__ out) {
    __shared__ float buf[64];
    const int head = blockIdx.x;
    const float v = logits[head * 64 + threadIdx.x];
    buf[threadIdx.x] = v; __syncthreads();
    for (int s = 32; s > 0; s >>= 1) {
        if (threadIdx.x < s) buf[threadIdx.x] = fmaxf(buf[threadIdx.x], buf[threadIdx.x + s]);
        __syncthreads();
    }
    const float m = buf[0]; __syncthreads();
    const float e = __expf(v - m);
    buf[threadIdx.x] = e; __syncthreads();
    for (int s = 32; s > 0; s >>= 1) {
        if (threadIdx.x < s) buf[threadIdx.x] += buf[threadIdx.x + s];
        __syncthreads();
    }
    out[head * 64 + threadIdx.x] = e / buf[0];
}

__global__ void __launch_bounds__(256)
value_partial(const float* __restrict__ hf, const float* __restrict__ wv,
              float* __restrict__ partials) {
    __shared__ float red[256];
    float a = 0.0f;
    for (int i = blockIdx.x * 256 + threadIdx.x; i < 131072; i += 128 * 256)
        a += hf[i] * wv[i];
    red[threadIdx.x] = a; __syncthreads();
    for (int s = 128; s > 0; s >>= 1) {
        if (threadIdx.x < s) red[threadIdx.x] += red[threadIdx.x + s];
        __syncthreads();
    }
    if (threadIdx.x == 0) partials[blockIdx.x] = red[0];
}

__global__ void value_final(const float* __restrict__ partials, const float* __restrict__ bv,
                            float* __restrict__ out) {
    __shared__ float red[128];
    red[threadIdx.x] = partials[threadIdx.x]; __syncthreads();
    for (int s = 64; s > 0; s >>= 1) {
        if (threadIdx.x < s) red[threadIdx.x] += red[threadIdx.x + s];
        __syncthreads();
    }
    if (threadIdx.x == 0) out[128] = red[0] + bv[0];
}

// ---------------------------------------------------------------------------
extern "C" void kernel_launch(void* const* d_in, const int* in_sizes, int n_in,
                              void* d_out, int out_size, void* d_ws, size_t ws_size,
                              hipStream_t stream) {
    const int*   states = (const int*)  d_in[0];
    const float* wemb   = (const float*)d_in[1];
    const float* bemb   = (const float*)d_in[2];
    const float* w1     = (const float*)d_in[3];
    const float* b1     = (const float*)d_in[4];
    const float* w2     = (const float*)d_in[5];
    const float* b2     = (const float*)d_in[6];
    const float* wx     = (const float*)d_in[7];
    const float* wh     = (const float*)d_in[8];
    const float* blstm  = (const float*)d_in[9];
    const float* wsp    = (const float*)d_in[10];
    const float* bs     = (const float*)d_in[11];
    const float* wpx    = (const float*)d_in[12];
    const float* bpx    = (const float*)d_in[13];
    const float* wpy    = (const float*)d_in[14];
    const float* bpy    = (const float*)d_in[15];
    const float* wv     = (const float*)d_in[16];
    const float* bv     = (const float*)d_in[17];
    float* out = (float*)d_out;

    // Workspace carve-out (~320 MB; dominated by the precomputed xz tensor).
    char* base = (char*)d_ws;
    size_t off = 0;
    auto alloc = [&](size_t bytes) -> char* {
        char* p = base + off;
        off += (bytes + 255) & ~(size_t)255;
        return p;
    };
    _Float16* x1   = (_Float16*)alloc((size_t)TSTEPS * 4096 * 16 * 2);
    _Float16* x2   = (_Float16*)alloc((size_t)TSTEPS * 4096 * 32 * 2);
    float*    xz   = (float*)   alloc((size_t)TSTEPS * 4096 * 128 * 4);
    float*    cbuf = (float*)   alloc((size_t)4096 * 32 * 4);
    float*    hf   = (float*)   alloc((size_t)4096 * 32 * 4);
    _Float16* hh0  = (_Float16*)alloc((size_t)4096 * 32 * 2);
    _Float16* hh1  = (_Float16*)alloc((size_t)4096 * 32 * 2);
    _Float16* w2p  = (_Float16*)alloc((size_t)32 * 160 * 2);
    _Float16* wxp  = (_Float16*)alloc((size_t)128 * 288 * 2);
    _Float16* whp  = (_Float16*)alloc((size_t)128 * 288 * 2);
    float*    sf   = (float*)   alloc(4096 * 4);
    float*    lg   = (float*)   alloc(128 * 4);
    float*    pt   = (float*)   alloc(128 * 4);

    init_pack<<<512, 256, 0, stream>>>(w2, wx, wh, w2p, wxp, whp, cbuf, hh0);
    embed_conv1<<<2048, 256, 0, stream>>>(states, wemb, bemb, w1, b1, x1);
    conv2_wmma<<<TSTEPS * 64, 256, 0, stream>>>(x1, w2p, b2, x2);
    convx_wmma<<<TSTEPS * 256, 256, 0, stream>>>(x2, wxp, blstm, xz);

    _Float16* hin = hh0;
    _Float16* hout = hh1;
    for (int t = 0; t < TSTEPS; ++t) {
        lstm_step<<<256, 256, 0, stream>>>(hin, whp, xz + (size_t)t * 4096 * 128,
                                           cbuf, hf, hout);
        _Float16* tmp = hin; hin = hout; hout = tmp;
    }

    spatial_head<<<16, 256, 0, stream>>>(hf, wsp, bs, sf);
    policy_gemv<<<128, 256, 0, stream>>>(sf, wpx, bpx, wpy, bpy, lg);
    softmax64<<<2, 64, 0, stream>>>(lg, out);
    value_partial<<<128, 256, 0, stream>>>(hf, wv, pt);
    value_final<<<1, 128, 0, stream>>>(pt, bv, out);
}
